// KreaCausalWanTransformerBlock_22342419874383
// MI455X (gfx1250) — matrix operand order repaired and checked by reading, not compiled, not run on hardware
//
#include <hip/hip_runtime.h>
#include <math.h>

#define FRAME_SEQLEN 1560
#define LOCAL_ATTN   4
#define NUM_HEADS_C  12

typedef __bf16 bf16_t;
typedef __attribute__((ext_vector_type(16))) __bf16 v16bf;
typedef __attribute__((ext_vector_type(8)))  __bf16 v8bf;
typedef __attribute__((ext_vector_type(8)))  float  v8f;
typedef __attribute__((ext_vector_type(4)))  unsigned int v4u;
typedef __attribute__((ext_vector_type(8)))  int v8i;
typedef __attribute__((ext_vector_type(4)))  int v4i;

#if __has_builtin(__builtin_amdgcn_tensor_load_to_lds) && __has_builtin(__builtin_amdgcn_s_wait_tensorcnt)
#define USE_TDM 1
#else
#define USE_TDM 0
#endif

// ---------------- helpers ----------------

__device__ __forceinline__ unsigned short f2bf_bits(float f) {
  union { float f; unsigned u; } x; x.f = f;
  unsigned u = x.u;
  u += 0x7fffu + ((u >> 16) & 1u);   // round to nearest even
  return (unsigned short)(u >> 16);
}
__device__ __forceinline__ bf16_t f2bf16(float f) {
  unsigned short s = f2bf_bits(f);
  return __builtin_bit_cast(bf16_t, s);
}

#if __has_builtin(__builtin_amdgcn_cvt_pk_bf16_f32)
typedef __attribute__((ext_vector_type(2))) __bf16 v2bf;
__device__ __forceinline__ void store_bf16_pair(unsigned short* p, float a, float b) {
  v2bf v = __builtin_amdgcn_cvt_pk_bf16_f32(a, b);
  *(v2bf*)p = v;
}
#else
__device__ __forceinline__ void store_bf16_pair(unsigned short* p, float a, float b) {
  p[0] = f2bf_bits(a); p[1] = f2bf_bits(b);
}
#endif

// A-operand fragment (16x32, lane = M row, chunked K per ISA table):
// lane<16:  elems 0..7 = K 0..7,  elems 8..15 = K 16..23
// lane>=16: elems 0..7 = K 8..15, elems 8..15 = K 24..31
__device__ __forceinline__ v16bf fragA_ld(const unsigned short* rowPtr, int lhalf) {
  const v8bf* p = (const v8bf*)(rowPtr + lhalf * 8);
  v8bf lo = p[0];
  v8bf hi = p[2];  // +16 halves
  v16bf r;
#pragma unroll
  for (int i = 0; i < 8; ++i) { r[i] = lo[i]; r[8 + i] = hi[i]; }
  return r;
}

// B-operand fragment (32x16): lane%16 = N col, 16 contiguous elems = K values of that col.
// Source LDS layout must be [N][K-contiguous]; p = &tile[n*rowStride + 16*h], k = 16h+s (A must match).
__device__ __forceinline__ v16bf fragB_ld(const unsigned short* p) {
  return *(const v16bf*)p;
}

__device__ __forceinline__ v8f wmma_bf16(v16bf a, v16bf b, v8f c) {
  return __builtin_amdgcn_wmma_f32_16x16x32_bf16(false, a, false, b, (short)0, c,
                                                 false, false);
}

#if USE_TDM
// Issue a TDM 2-D load: tile of [rows=nrows][cols=ncols] bf16 elements from
// global (row stride = strideElems) into LDS at ldsAddr, natural layout.
// tensorRows < nrows rows => hardware OOB zero-fill.
__device__ __forceinline__ void tdm_load_2d_bf16(unsigned ldsAddr,
                                                 const unsigned short* gptr,
                                                 int ncols, int nrows,
                                                 int tensorRows, int strideElems) {
  unsigned long long ga = (unsigned long long)gptr;
  v4u g0;
  g0[0] = 1u;                                        // count=1, user descriptor
  g0[1] = ldsAddr;                                   // lds_addr [63:32]
  g0[2] = (unsigned)(ga & 0xffffffffu);              // global_addr low
  g0[3] = (unsigned)((ga >> 32) & 0x01ffffffu)       // global_addr [56:32]
        | (2u << 30);                                // type = 2 (image)
  v8i g1;
  g1[0] = (1 << 16);                                 // data_size = 1 (2 bytes)
  g1[1] = (int)((unsigned)(ncols & 0xffff) << 16);   // tensor_dim0[15:0]
  g1[2] = (int)((unsigned)(tensorRows & 0xffff) << 16); // td0 hi=0 | tensor_dim1[15:0]
  g1[3] = (int)((unsigned)(ncols & 0xffff) << 16);   // td1 hi=0 | tile_dim0
  g1[4] = nrows & 0xffff;                            // tile_dim1 | tile_dim2=0
  g1[5] = strideElems;                               // tensor_dim0_stride low 32
  g1[6] = 0;                                         // stride hi | dim1_stride lo
  g1[7] = 0;
  v4i gz = {0, 0, 0, 0};
  v8i gz8 = {0, 0, 0, 0, 0, 0, 0, 0};
  __builtin_amdgcn_tensor_load_to_lds(g0, g1, gz, gz, gz8, 0);
}
#endif

// ---------------- GEMM: C[M,N] = A[M,K] @ W[N,K]^T (+bias) ----------------
// EPI 0: out = v ; EPI 1: out = gelu_tanh(v) ; EPI 2: out += v * gate
template <int EPI>
__global__ __launch_bounds__(256) void gemm_kernel(
    const float* __restrict__ A, const float* __restrict__ W,
    const float* __restrict__ bias, float* __restrict__ out,
    int M, int N, int K,
    const float* __restrict__ sst, const float* __restrict__ temb,
    int gidx, int fs) {
  __shared__ __align__(32) unsigned short As[128 * 32];   // [m][k] (A layout)
  __shared__ __align__(32) unsigned short Bst[32 * 128];  // [n][k] (W layout, read like A)
  const int tid = threadIdx.x;
  const int lane = tid & 31, wid = tid >> 5;
  const int wm = wid & 3, wn = wid >> 2;       // 4 x 2 wave grid
  const int lhalf = lane >> 4, lrow = lane & 15;
  const int bm = blockIdx.y * 128, bn = blockIdx.x * 128;

  v8f acc[2][4] = {};

  for (int k0 = 0; k0 < K; k0 += 32) {
#pragma unroll
    for (int i = 0; i < 4; ++i) {
      int g = tid + i * 256;           // 1024 float4 groups per tile
      int r = g >> 3;
      int c4 = (g & 7) << 2;
      int gr = bm + r;
      float4 av = make_float4(0.f, 0.f, 0.f, 0.f);
      if (gr < M) av = *(const float4*)(A + (size_t)gr * K + k0 + c4);
      unsigned short* da = &As[r * 32 + c4];
      store_bf16_pair(da, av.x, av.y);
      store_bf16_pair(da + 2, av.z, av.w);
      float4 wv = *(const float4*)(W + (size_t)(bn + r) * K + k0 + c4);
      // natural store: Bst[n][k]
      Bst[r * 32 + c4 + 0] = f2bf_bits(wv.x);
      Bst[r * 32 + c4 + 1] = f2bf_bits(wv.y);
      Bst[r * 32 + c4 + 2] = f2bf_bits(wv.z);
      Bst[r * 32 + c4 + 3] = f2bf_bits(wv.w);
    }
    if (k0 + 32 < K) {                 // CDNA5 global_prefetch_b8 of next tile
      int pr = tid >> 1;
      if (bm + pr < M) __builtin_prefetch(A + (size_t)(bm + pr) * K + k0 + 32, 0, 1);
      __builtin_prefetch(W + (size_t)(bn + pr) * K + k0 + 32, 0, 1);
    }
    __syncthreads();

    v16bf af[2], bfm[4];
#pragma unroll
    for (int mt = 0; mt < 2; ++mt)
      af[mt] = fragA_ld(&As[(wm * 32 + mt * 16 + lrow) * 32], lhalf);
#pragma unroll
    for (int nt = 0; nt < 4; ++nt)
      bfm[nt] = fragA_ld(&Bst[(wn * 64 + nt * 16 + lrow) * 32], lhalf);
#pragma unroll
    for (int mt = 0; mt < 2; ++mt)
#pragma unroll
      for (int nt = 0; nt < 4; ++nt)
        acc[mt][nt] = wmma_bf16(af[mt], bfm[nt], acc[mt][nt]);
    __syncthreads();
  }

  const int rbase = lhalf * 8;
#pragma unroll
  for (int mt = 0; mt < 2; ++mt)
#pragma unroll
    for (int nt = 0; nt < 4; ++nt)
#pragma unroll
      for (int j = 0; j < 8; ++j) {
        int gr = bm + wm * 32 + mt * 16 + rbase + j;
        int gc = bn + wn * 64 + nt * 16 + lrow;
        if (gr < M) {
          float v = acc[mt][nt][j];
          if (bias) v += bias[gc];
          if (EPI == 1) {
            float t = 0.7978845608028654f * (v + 0.044715f * v * v * v);
            v = 0.5f * v * (1.0f + tanhf(t));
          }
          if (EPI == 2) {
            float g = 1.0f;
            if (gidx >= 0) {
              int fr = gr / fs;
              g = sst[(size_t)gidx * N + gc] +
                  temb[((size_t)fr * 6 + gidx) * N + gc];
            }
            out[(size_t)gr * N + gc] += v * g;
          } else {
            out[(size_t)gr * N + gc] = v;
          }
        }
      }
}

// ---------------- LayerNorm (+AdaLN modulation or affine) ----------------
__global__ __launch_bounds__(256) void ln_mod_kernel(
    const float* __restrict__ x, float* __restrict__ out, int D,
    const float* __restrict__ w, const float* __restrict__ b,
    const float* __restrict__ sst, const float* __restrict__ temb,
    int scaleIdx, int shiftIdx, int fs) {
  __shared__ float red[256];
  const int l = blockIdx.x, tid = threadIdx.x;
  const float* xr = x + (size_t)l * D;
  float s = 0.f, s2 = 0.f;
  for (int d = tid; d < D; d += 256) { float v = xr[d]; s += v; s2 += v * v; }
  red[tid] = s; __syncthreads();
  for (int st = 128; st > 0; st >>= 1) { if (tid < st) red[tid] += red[tid + st]; __syncthreads(); }
  float mean = red[0] / D; __syncthreads();
  red[tid] = s2; __syncthreads();
  for (int st = 128; st > 0; st >>= 1) { if (tid < st) red[tid] += red[tid + st]; __syncthreads(); }
  float var = red[0] / D - mean * mean;
  float inv = rsqrtf(var + 1e-6f);
  int fr = l / fs;
  float* orow = out + (size_t)l * D;
  for (int d = tid; d < D; d += 256) {
    float y = (xr[d] - mean) * inv;
    if (scaleIdx >= 0) {
      float sc = 1.f + sst[(size_t)scaleIdx * D + d] + temb[((size_t)fr * 6 + scaleIdx) * D + d];
      float sh = sst[(size_t)shiftIdx * D + d] + temb[((size_t)fr * 6 + shiftIdx) * D + d];
      y = y * sc + sh;
    } else if (w) {
      y = y * w[d] + b[d];
    }
    orow[d] = y;
  }
}

// ---------------- RMSNorm (in place, rowwise over D) ----------------
__global__ __launch_bounds__(256) void rms_kernel(float* __restrict__ x,
                                                  const float* __restrict__ w, int D) {
  __shared__ float red[256];
  const int l = blockIdx.x, tid = threadIdx.x;
  float* xr = x + (size_t)l * D;
  float s2 = 0.f;
  for (int d = tid; d < D; d += 256) { float v = xr[d]; s2 += v * v; }
  red[tid] = s2; __syncthreads();
  for (int st = 128; st > 0; st >>= 1) { if (tid < st) red[tid] += red[tid + st]; __syncthreads(); }
  float inv = rsqrtf(red[0] / D + 1e-6f);
  for (int d = tid; d < D; d += 256) xr[d] = xr[d] * inv * w[d];
}

// ---------------- RoPE (in place on q and k) ----------------
__global__ __launch_bounds__(256) void rope_kernel(
    float* __restrict__ q, float* __restrict__ k, int H, int hd,
    const float* __restrict__ fre, const float* __restrict__ fim,
    const int* __restrict__ d_cs, const int* __restrict__ d_gh,
    const int* __restrict__ d_gw) {
  const int l = blockIdx.x, tid = threadIdx.x;
  const int gh = *d_gh, gw = *d_gw;
  const int fsz = gh * gw;
  const int start_frame = (*d_cs) / FRAME_SEQLEN;
  const int f = l / fsz, rem = l % fsz;
  const int hh = rem / gw, ww = rem % gw;
  const int c = hd >> 1;
  const int s0 = c - 2 * (c / 3), s1 = c / 3;
  const int total = H * c;
  for (int t = tid; t < total; t += 256) {
    int hidx = t / c, j = t % c;
    int row = (j < s0) ? (start_frame + f) : ((j < s0 + s1) ? hh : ww);
    float cv = fre[(size_t)row * c + j];
    float sv = fim[(size_t)row * c + j];
    size_t base = ((size_t)l * H + hidx) * (size_t)hd + 2 * j;
    float a = q[base], b = q[base + 1];
    q[base] = a * cv - b * sv; q[base + 1] = a * sv + b * cv;
    a = k[base]; b = k[base + 1];
    k[base] = a * cv - b * sv; k[base + 1] = a * sv + b * cv;
  }
}

// ---------------- build bf16 KV window from cache + new tokens ----------------
__global__ void build_kv_kernel(const float* __restrict__ kc, const float* __restrict__ vc,
                                const float* __restrict__ nk, const float* __restrict__ nv,
                                unsigned short* __restrict__ Kw, unsigned short* __restrict__ Vw,
                                int L, int D, const int* __restrict__ d_cs) {
  const int cs = *d_cs;
  const int local_end = cs + L;
  int win_lo = local_end - LOCAL_ATTN * FRAME_SEQLEN; if (win_lo < 0) win_lo = 0;
  const int kvLen = local_end - win_lo;
  const int nOld = cs - win_lo;
  const size_t total = (size_t)kvLen * D;
  const size_t stride = (size_t)gridDim.x * blockDim.x;
  for (size_t i = (size_t)blockIdx.x * blockDim.x + threadIdx.x; i < total; i += stride) {
    int t = (int)(i / D), d = (int)(i % D);
    float kvv, vvv;
    if (t < nOld) {
      kvv = kc[(size_t)(win_lo + t) * D + d];
      vvv = vc[(size_t)(win_lo + t) * D + d];
    } else {
      kvv = nk[(size_t)(t - nOld) * D + d];
      vvv = nv[(size_t)(t - nOld) * D + d];
    }
    Kw[i] = f2bf_bits(kvv);
    Vw[i] = f2bf_bits(vvv);
  }
}

__global__ void cast_bf16_kernel(const float* __restrict__ src,
                                 unsigned short* __restrict__ dst, size_t n) {
  size_t i = (size_t)blockIdx.x * blockDim.x + threadIdx.x;
  const size_t stride = (size_t)gridDim.x * blockDim.x;
  for (; i < n; i += stride) dst[i] = f2bf_bits(src[i]);
}

// ---------------- flash attention: 1 wave = 16 q-rows x 1 head ----------------
__global__ __launch_bounds__(32) void attn_kernel(
    const float* __restrict__ Q, const unsigned short* __restrict__ Kw,
    const unsigned short* __restrict__ Vw, float* __restrict__ O,
    int Lq, int D, int H, int hd, int kvFixed, const int* __restrict__ d_cs, int Lnew) {
  __shared__ __align__(32) unsigned short KtT[128 * 32];  // [tok][hd]  (B source for S)
  __shared__ __align__(32) unsigned short Vt[32 * 128];   // [tok][hd]  (B layout for O)
  __shared__ __align__(32) unsigned short Pt[16 * 32];    // [qrow][tok]
  const int lane = threadIdx.x;
  const int lhalf = lane >> 4, lrow = lane & 15;
  const int q0 = blockIdx.x * 16;
  const int h = blockIdx.y;
  int kvLen;
  if (d_cs) {
    int cs = *d_cs;
    int local_end = cs + Lnew;
    int win_lo = local_end - LOCAL_ATTN * FRAME_SEQLEN; if (win_lo < 0) win_lo = 0;
    kvLen = local_end - win_lo;
  } else {
    kvLen = kvFixed;
  }
  const float scale = rsqrtf((float)hd);

  // Q fragments held in registers (4 x 16x32 bf16 over hd=128), A layout
  v16bf qf[4];
  {
    int qr = q0 + lrow; if (qr >= Lq) qr = Lq - 1;
    const float* qrow = Q + (size_t)qr * D + (size_t)h * hd;
#pragma unroll
    for (int c = 0; c < 4; ++c) {
      const float* p = qrow + c * 32 + lhalf * 8;
      v16bf r;
#pragma unroll
      for (int i = 0; i < 8; ++i) { r[i] = f2bf16(p[i]); r[8 + i] = f2bf16(p[16 + i]); }
      qf[c] = r;
    }
  }

  float mrun[8], lrun[8];
  v8f oacc[8] = {};
#pragma unroll
  for (int j = 0; j < 8; ++j) { mrun[j] = -3.0e38f; lrun[j] = 0.f; }

  for (int kt = 0; kt < kvLen; kt += 32) {
    const int gk = kt + lane;
    const bool okk = (gk < kvLen);
    const unsigned short* krow = Kw + (size_t)gk * D + (size_t)h * hd;
    // K tile: store into [tok][hd]
#pragma unroll
    for (int d0 = 0; d0 < 128; d0 += 8) {
      union { uint4 v; unsigned short s[8]; } u;
      u.v = okk ? *(const uint4*)(krow + d0) : make_uint4(0, 0, 0, 0);
#pragma unroll
      for (int j = 0; j < 8; ++j) KtT[lane * 128 + d0 + j] = u.s[j];
    }
#if USE_TDM
    {
      int remTok = kvLen - kt;                      // OOB rows auto zero-fill
      tdm_load_2d_bf16((unsigned)(unsigned long long)&Vt[0],
                       Vw + (size_t)kt * D + (size_t)h * hd,
                       /*ncols=*/128, /*nrows=*/32, /*tensorRows=*/remTok,
                       /*strideElems=*/D);
    }
    __builtin_amdgcn_s_wait_tensorcnt(0);
#else
    {
      const unsigned short* vrow = Vw + (size_t)gk * D + (size_t)h * hd;
#pragma unroll
      for (int d0 = 0; d0 < 128; d0 += 8) {
        uint4 v4 = okk ? *(const uint4*)(vrow + d0) : make_uint4(0, 0, 0, 0);
        *(uint4*)&Vt[lane * 128 + d0] = v4;
      }
    }
#endif
    __syncthreads();

    // S = Q @ K^T for this 16x32 tile; B-frag: lane%16 = token, elems = hd (table order)
    v8f sfr[2] = {};
#pragma unroll
    for (int f_ = 0; f_ < 2; ++f_)
#pragma unroll
      for (int c = 0; c < 4; ++c) {
        v16bf bk = fragA_ld(&KtT[(f_ * 16 + lrow) * 128 + c * 32], lhalf);
        sfr[f_] = wmma_bf16(qf[c], bk, sfr[f_]);
      }

    // online softmax update (16-lane group <-> one q row of the C fragment)
#pragma unroll
    for (int j = 0; j < 8; ++j) {
      float v0 = sfr[0][j] * scale, v1 = sfr[1][j] * scale;
      const int c0 = kt + lrow, c1 = c0 + 16;
      if (c0 >= kvLen) v0 = -3.0e38f;
      if (c1 >= kvLen) v1 = -3.0e38f;
      float t = fmaxf(v0, v1);
#pragma unroll
      for (int m = 1; m < 16; m <<= 1) t = fmaxf(t, __shfl_xor(t, m, 32));
      float newm = fmaxf(mrun[j], t);
      float corr = __expf(mrun[j] - newm);
      mrun[j] = newm;
      v0 = (c0 < kvLen) ? __expf(v0 - newm) : 0.f;
      v1 = (c1 < kvLen) ? __expf(v1 - newm) : 0.f;
      float sum = v0 + v1;
#pragma unroll
      for (int m = 1; m < 16; m <<= 1) sum += __shfl_xor(sum, m, 32);
      lrun[j] = lrun[j] * corr + sum;
#pragma unroll
      for (int nt = 0; nt < 8; ++nt) oacc[nt][j] *= corr;
      Pt[(lhalf * 8 + j) * 32 + lrow] = f2bf_bits(v0);
      Pt[(lhalf * 8 + j) * 32 + lrow + 16] = f2bf_bits(v1);
    }
    __syncthreads();

    // O += P @ V ; A-frag from Pt, B-frag: lane%16 = hd col, elems = tokens (table order)
    v16bf pf = fragA_ld(&Pt[lrow * 32], lhalf);
#pragma unroll
    for (int nt = 0; nt < 8; ++nt) {
      v16bf bv;
#pragma unroll
      for (int i = 0; i < 8; ++i) {
        bv[i] = __builtin_bit_cast(bf16_t, Vt[(lhalf * 8 + i) * 128 + nt * 16 + lrow]);
        bv[8 + i] = __builtin_bit_cast(bf16_t, Vt[(16 + lhalf * 8 + i) * 128 + nt * 16 + lrow]);
      }
      oacc[nt] = wmma_bf16(pf, bv, oacc[nt]);
    }
    __syncthreads();
  }

#pragma unroll
  for (int nt = 0; nt < 8; ++nt)
#pragma unroll
    for (int j = 0; j < 8; ++j) {
      int gr = q0 + lhalf * 8 + j;
      if (gr < Lq) {
        float val = oacc[nt][j] / lrun[j];
        O[(size_t)gr * D + (size_t)h * hd + nt * 16 + lrow] = val;
      }
    }
}

// ---------------- host orchestration ----------------

static inline size_t alignup(size_t x) { return (x + 255) & ~(size_t)255; }

extern "C" void kernel_launch(void* const* d_in, const int* in_sizes, int n_in,
                              void* d_out, int out_size, void* d_ws, size_t ws_size,
                              hipStream_t stream) {
  const float* hid   = (const float*)d_in[0];
  const float* ctx   = (const float*)d_in[1];
  const float* temb  = (const float*)d_in[2];
  const float* fre   = (const float*)d_in[3];
  const float* fim   = (const float*)d_in[4];
  const float* kcache= (const float*)d_in[5];
  const float* vcache= (const float*)d_in[6];
  const float* sst   = (const float*)d_in[7];
  const float* Wq = (const float*)d_in[8];  const float* bq = (const float*)d_in[9];
  const float* Wk = (const float*)d_in[10]; const float* bk = (const float*)d_in[11];
  const float* Wv = (const float*)d_in[12]; const float* bv = (const float*)d_in[13];
  const float* Wo = (const float*)d_in[14]; const float* bo = (const float*)d_in[15];
  const float* norm_q_w = (const float*)d_in[16];
  const float* norm_k_w = (const float*)d_in[17];
  const float* saln_w = (const float*)d_in[18];
  const float* saln_b = (const float*)d_in[19];
  const float* cWq = (const float*)d_in[20]; const float* cbq = (const float*)d_in[21];
  const float* cWk = (const float*)d_in[22]; const float* cbk = (const float*)d_in[23];
  const float* cWv = (const float*)d_in[24]; const float* cbv = (const float*)d_in[25];
  const float* cWo = (const float*)d_in[26]; const float* cbo = (const float*)d_in[27];
  const float* c_norm_q_w = (const float*)d_in[28];
  const float* c_norm_k_w = (const float*)d_in[29];
  const float* W1 = (const float*)d_in[30]; const float* b1 = (const float*)d_in[31];
  const float* W2 = (const float*)d_in[32]; const float* b2 = (const float*)d_in[33];
  const int* d_cs = (const int*)d_in[34];
  const int* d_gh = (const int*)d_in[35];
  const int* d_gw = (const int*)d_in[36];

  const int D = (int)(sqrt((double)in_sizes[8]) + 0.5);
  const int L = in_sizes[0] / D;
  const int CTXN = in_sizes[1] / D;
  const int NF = in_sizes[2] / (6 * D);
  const int FF = in_sizes[30] / D;
  const int CACHE = in_sizes[5] / D;
  const int fs = L / NF;
  const int H = NUM_HEADS_C;
  const int hd = D / H;

  const size_t LD4 = (size_t)L * D * sizeof(float);
  const size_t KVB = (size_t)CACHE * D * sizeof(unsigned short);
  char* w = (char*)d_ws;
  const size_t off1 = alignup(LD4);
  float* bufA = (float*)(w + 0);
  float* bufQ = (float*)(w + off1);
  float* bufK = (float*)(w + off1 + alignup(LD4));
  float* bufV = (float*)(w + off1 + 2 * alignup(LD4));
  unsigned short* Kw = (unsigned short*)(w + off1 + 3 * alignup(LD4));
  unsigned short* Vw = (unsigned short*)(w + off1 + 3 * alignup(LD4) + alignup(KVB));
  float* Ff = (float*)(w + off1);  // aliases Q/K/V/Kw/Vw region; only live in FFN phase
  float* xo = (float*)d_out;

  auto gemm = [&](int epi, const float* Ap, const float* Wp, const float* bp,
                  float* op, int M, int N, int K, int gidx) {
    dim3 g(N / 128, (M + 127) / 128), b(256);
    if (epi == 0)
      gemm_kernel<0><<<g, b, 0, stream>>>(Ap, Wp, bp, op, M, N, K, sst, temb, gidx, fs);
    else if (epi == 1)
      gemm_kernel<1><<<g, b, 0, stream>>>(Ap, Wp, bp, op, M, N, K, sst, temb, gidx, fs);
    else
      gemm_kernel<2><<<g, b, 0, stream>>>(Ap, Wp, bp, op, M, N, K, sst, temb, gidx, fs);
  };

  // x := hidden_states (running residual lives in d_out)
  (void)hipMemcpyAsync(xo, hid, LD4, hipMemcpyDeviceToDevice, stream);

  // nh = LN(x)*(1+scale)+shift  (AdaLN indices: scale=1, shift=0)
  ln_mod_kernel<<<L, 256, 0, stream>>>(xo, bufA, D, nullptr, nullptr, sst, temb, 1, 0, fs);

  // q,k,v projections + RMS + RoPE
  gemm(0, bufA, Wq, bq, bufQ, L, D, D, -1);
  rms_kernel<<<L, 256, 0, stream>>>(bufQ, norm_q_w, D);
  gemm(0, bufA, Wk, bk, bufK, L, D, D, -1);
  rms_kernel<<<L, 256, 0, stream>>>(bufK, norm_k_w, D);
  gemm(0, bufA, Wv, bv, bufV, L, D, D, -1);
  rope_kernel<<<L, 256, 0, stream>>>(bufQ, bufK, H, hd, fre, fim, d_cs, d_gh, d_gw);

  // KV window (cache tail + new tokens) in bf16
  build_kv_kernel<<<2048, 256, 0, stream>>>(kcache, vcache, bufK, bufV, Kw, Vw, L, D, d_cs);

  // self attention -> bufA ; x += (attn @ Wo + bo) * gate(idx 2)
  attn_kernel<<<dim3((L + 15) / 16, H), 32, 0, stream>>>(bufQ, Kw, Vw, bufA, L, D, H, hd, 0, d_cs, L);
  gemm(2, bufA, Wo, bo, xo, L, D, D, 2);

  // cross attention
  ln_mod_kernel<<<L, 256, 0, stream>>>(xo, bufA, D, saln_w, saln_b, sst, temb, -1, -1, fs);
  gemm(0, bufA, cWq, cbq, bufQ, L, D, D, -1);
  rms_kernel<<<L, 256, 0, stream>>>(bufQ, c_norm_q_w, D);
  gemm(0, ctx, cWk, cbk, bufV, CTXN, D, D, -1);
  rms_kernel<<<CTXN, 256, 0, stream>>>(bufV, c_norm_k_w, D);
  cast_bf16_kernel<<<1024, 256, 0, stream>>>(bufV, Kw, (size_t)CTXN * D);
  gemm(0, ctx, cWv, cbv, bufV, CTXN, D, D, -1);
  cast_bf16_kernel<<<1024, 256, 0, stream>>>(bufV, Vw, (size_t)CTXN * D);
  attn_kernel<<<dim3((L + 15) / 16, H), 32, 0, stream>>>(bufQ, Kw, Vw, bufK, L, D, H, hd, CTXN, nullptr, L);
  gemm(2, bufK, cWo, cbo, xo, L, D, D, -1);  // x += cross @ cWo + cbo (no gate)

  // FFN: n3 = LN(x)*(1+c_scale)+c_shift (scale=4, shift=3); x += gelu(n3@W1)@W2 * c_gate(5)
  ln_mod_kernel<<<L, 256, 0, stream>>>(xo, bufA, D, nullptr, nullptr, sst, temb, 4, 3, fs);
  gemm(1, bufA, W1, b1, Ff, L, FF, D, -1);
  gemm(2, Ff, W2, b2, xo, L, D, FF, 5);
}
